// HIPPIEncoder_27917287424800
// MI455X (gfx1250) — matrix operand order repaired and checked by reading, not compiled, not run on hardware
//
#include <hip/hip_runtime.h>
#include <math.h>

#define N_NODES 6144
#define IN_DIM  512
#define HYPER   256
#define D0      513
#define KPAD_L  544      // D0 padded up to multiple of 32
#define C_CLASS 7
#define N_EDGES (N_NODES*32)
#define GCOLS   272      // 256 (scores*xt) + 1 (scores) col, padded to 16
#define GROWS_PAD 320    // Gt rows padded to multiple of 64 for unguarded staging
#define OUT_COLS 2305    // 513 + 7*256
#define EPSC 0.004f
#define MINN 1e-15f

typedef float  v8f   __attribute__((ext_vector_type(8)));
typedef __bf16 v16bf __attribute__((ext_vector_type(16)));
typedef __bf16 v8bf  __attribute__((ext_vector_type(8)));
typedef __bf16 v4bf  __attribute__((ext_vector_type(4)));
typedef int    v4i   __attribute__((vector_size(4 * sizeof(int))));

// hardware f32 -> bf16 (v_cvt_*bf16_f32), RNE
__device__ __forceinline__ __bf16 f2bf(float f) { return (__bf16)f; }

// ---------------------------------------------------------------------------
// CDNA5 async global->LDS 16-byte copy (ASYNCcnt-tracked DMA, no VGPR data).
// Toolchain signature (from probe): (v4i AS1* src, v4i AS3* dst, imm, cpol).
// ---------------------------------------------------------------------------
#if defined(__has_builtin)
#if __has_builtin(__builtin_amdgcn_global_load_async_to_lds_b128)
#define HAVE_ASYNC_LDS 1
#endif
#endif

__device__ __forceinline__ void copy_b128_to_lds(const void* g, void* l) {
#ifdef HAVE_ASYNC_LDS
  typedef __attribute__((address_space(1))) v4i* gp_t;   // global (AS1)
  typedef __attribute__((address_space(3))) v4i* lp_t;   // LDS (AS3)
  gp_t gp = (gp_t)(unsigned long long)(uintptr_t)g;      // generic->AS1 numeric
  lp_t lp = (lp_t)(unsigned int)(uintptr_t)l;            // generic->AS3 low 32b
  __builtin_amdgcn_global_load_async_to_lds_b128(gp, lp, 0, 0);
#else
  *(float4*)l = *(const float4*)g;
#endif
}

__device__ __forceinline__ void async_lds_wait() {
#ifdef HAVE_ASYNC_LDS
#if __has_builtin(__builtin_amdgcn_s_wait_asynccnt)
  __builtin_amdgcn_s_wait_asynccnt(0);
#else
  asm volatile("s_wait_asynccnt 0x0" ::: "memory");
#endif
#endif
}

// 256-thread block sum; all threads must participate.
__device__ __forceinline__ float blockSum(float v, float* red) {
  int t = threadIdx.x;
  red[t] = v; __syncthreads();
  #pragma unroll
  for (int s = 128; s > 0; s >>= 1) {
    if (t < s) red[t] += red[t + s];
    __syncthreads();
  }
  float r = red[0]; __syncthreads();
  return r;
}

// ---------------------------------------------------------------------------
// adj (f32, 151MB) -> adjb (bf16, 75.5MB): done ONCE; bf16 copy is L2-resident
// across all 7 attention GEMMs (192MB L2), halving L2 bandwidth per pass.
// ---------------------------------------------------------------------------
__global__ __launch_bounds__(256)
void cvt_adj_kernel(const float* __restrict__ a, __bf16* __restrict__ b)
{
  size_t i = ((size_t)blockIdx.x * 256 + threadIdx.x) * 4;
  float4 v = *(const float4*)(a + i);
  v4bf o = { f2bf(v.x), f2bf(v.y), f2bf(v.z), f2bf(v.w) };
  *(v4bf*)(b + i) = o;
}

// ---------------------------------------------------------------------------
// Big GEMM (attention aggregation): C[M x N] = A(bf16,[m][k]) * B(bf16,[n][k])^T
// Block tile 128x64, 8 waves, each wave 32x32 of C via 4 independent
// v_wmma_f32_16x16x32_bf16 per K-step. Double-buffered LDS; tile k+1 is
// DMA'd (async-to-LDS) while tile k is multiplied. M%128==0, K%32==0,
// B must be readable for 64-aligned row range (Gt padded to 320 rows).
// ---------------------------------------------------------------------------
__global__ __launch_bounds__(256)
void gemm_wmma_bf16bf16(const __bf16* __restrict__ A, int lda,
                        const __bf16* __restrict__ B, int ldb,
                        float* __restrict__ C, int ldc,
                        int N, int K)
{
  __shared__ __align__(16) __bf16 As[2][128 * 40];
  __shared__ __align__(16) __bf16 Bs[2][64 * 40];

  const int tid  = threadIdx.x;
  const int lane = tid & 31;
  const int wave = tid >> 5;
  const int wm   = wave >> 1;          // 0..3 : 32-row slices of M
  const int wn   = wave & 1;           // 0..1 : 32-col slices of N
  const int bm   = blockIdx.y * 128;
  const int bn   = blockIdx.x * 64;
  const int arow = tid >> 2;           // 0..63
  const int akq  = tid & 3;            // k-chunk of 8 bf16 (16 bytes)

  v8f acc00 = {}, acc01 = {}, acc10 = {}, acc11 = {};

  auto stage = [&](int k0, int buf) {
    const __bf16* ga0 = A + (size_t)(bm + arow) * lda + k0 + akq * 8;
    const __bf16* ga1 = ga0 + (size_t)64 * lda;
    const __bf16* gb  = B + (size_t)(bn + arow) * ldb + k0 + akq * 8;
    copy_b128_to_lds(ga0, &As[buf][arow * 40 + akq * 8]);
    copy_b128_to_lds(ga1, &As[buf][(arow + 64) * 40 + akq * 8]);
    copy_b128_to_lds(gb,  &Bs[buf][arow * 40 + akq * 8]);
  };

  const int nk = K >> 5;
  stage(0, 0);
  async_lds_wait();
  __syncthreads();

  for (int kt = 0; kt < nk; ++kt) {
    const int cur = kt & 1;
    if (kt + 1 < nk) stage((kt + 1) << 5, cur ^ 1);   // overlap DMA with WMMA

    const __bf16* Ab = As[cur];
    const __bf16* Bb = Bs[cur];
    // A 16x32 frag: lane l: m=l%16, koff=(l/16)*8; VGPR0-3 K=koff.., VGPR4-7 K=16+koff..
    int mrow = wm * 32 + (lane & 15);
    int koff = (lane >> 4) * 8;
    union Frag { v16bf v; v8bf h[2]; };
    Frag a0, a1, b0, b1;
    a0.h[0] = *(const v8bf*)&Ab[mrow * 40 + koff];
    a0.h[1] = *(const v8bf*)&Ab[mrow * 40 + 16 + koff];
    a1.h[0] = *(const v8bf*)&Ab[(mrow + 16) * 40 + koff];
    a1.h[1] = *(const v8bf*)&Ab[(mrow + 16) * 40 + 16 + koff];
    // B 32x16 frag: lane l: n=l%16, kb=(l/16)*8; VGPR0-3 K=kb.., VGPR4-7 K=16+kb..
    int nrow = wn * 32 + (lane & 15);
    int kb = (lane >> 4) * 8;
    b0.h[0] = *(const v8bf*)&Bb[nrow * 40 + kb];
    b0.h[1] = *(const v8bf*)&Bb[nrow * 40 + kb + 16];
    b1.h[0] = *(const v8bf*)&Bb[(nrow + 16) * 40 + kb];
    b1.h[1] = *(const v8bf*)&Bb[(nrow + 16) * 40 + kb + 16];

    acc00 = __builtin_amdgcn_wmma_f32_16x16x32_bf16(false, a0.v, false, b0.v, (short)0, acc00, false, false);
    acc01 = __builtin_amdgcn_wmma_f32_16x16x32_bf16(false, a0.v, false, b1.v, (short)0, acc01, false, false);
    acc10 = __builtin_amdgcn_wmma_f32_16x16x32_bf16(false, a1.v, false, b0.v, (short)0, acc10, false, false);
    acc11 = __builtin_amdgcn_wmma_f32_16x16x32_bf16(false, a1.v, false, b1.v, (short)0, acc11, false, false);

    async_lds_wait();
    __syncthreads();
  }

  // C/D layout: lane l: n=l%16; VGPR r: m = r + 8*(l/16)
  int crow = bm + wm * 32 + (lane >> 4) * 8;
  int col0 = bn + wn * 32 + (lane & 15);
  int col1 = col0 + 16;
  #pragma unroll
  for (int r = 0; r < 8; r++) {
    if (col0 < N) C[(size_t)(crow + r) * ldc + col0] = acc00[r];
    if (col1 < N) C[(size_t)(crow + r) * ldc + col1] = acc01[r];
    if (col0 < N) C[(size_t)(crow + 16 + r) * ldc + col0] = acc10[r];
    if (col1 < N) C[(size_t)(crow + 16 + r) * ldc + col1] = acc11[r];
  }
}

// ---------------------------------------------------------------------------
// Weight GEMM: C = A(f32,[m][k]) * W(f32,[n][k])^T + bias. f32 staged to bf16
// in LDS with hardware converts. Block tile 64x64, wave -> 16x32 of C.
// ---------------------------------------------------------------------------
__global__ __launch_bounds__(256)
void gemm_wmma_f32in(const float* __restrict__ A, int lda,
                     const float* __restrict__ B, int ldb,
                     const float* __restrict__ bias,
                     float* __restrict__ C, int ldc,
                     int N, int Kpad, int Kact)
{
  __shared__ __align__(16) __bf16 As[64 * 40];
  __shared__ __align__(16) __bf16 Bs[64 * 40];

  const int tid  = threadIdx.x;
  const int lane = tid & 31;
  const int wave = tid >> 5;
  const int wm   = wave >> 1;
  const int wn   = wave & 1;
  const int bm   = blockIdx.y * 64;
  const int bn   = blockIdx.x * 64;

  v8f acc0 = {}, acc1 = {};

  for (int k0 = 0; k0 < Kpad; k0 += 32) {
    #pragma unroll
    for (int i = 0; i < 2; i++) {
      int lin = tid + i * 256;
      int m = lin >> 3, kq = lin & 7;
      const float4 av = *(const float4*)(A + (size_t)(bm + m) * lda + k0 + kq * 4);
      v4bf o = { f2bf(av.x), f2bf(av.y), f2bf(av.z), f2bf(av.w) };
      *(v4bf*)&As[m * 40 + kq * 4] = o;
    }
    #pragma unroll
    for (int i = 0; i < 2; i++) {
      int lin = tid + i * 256;
      int n = lin >> 3, kq = lin & 7;
      int ng = bn + n;
      float b0 = 0.f, b1 = 0.f, b2 = 0.f, b3 = 0.f;
      if (ng < N) {
        int kk = k0 + kq * 4;
        const float* bp = B + (size_t)ng * ldb;
        if (kk + 0 < Kact) b0 = bp[kk + 0];
        if (kk + 1 < Kact) b1 = bp[kk + 1];
        if (kk + 2 < Kact) b2 = bp[kk + 2];
        if (kk + 3 < Kact) b3 = bp[kk + 3];
      }
      v4bf o = { f2bf(b0), f2bf(b1), f2bf(b2), f2bf(b3) };
      *(v4bf*)&Bs[n * 40 + kq * 4] = o;
    }
    __syncthreads();

    int mrow = wm * 16 + (lane & 15);
    int koff = (lane >> 4) * 8;
    union Frag { v16bf v; v8bf h[2]; };
    Frag af, bf0, bf1;
    af.h[0] = *(const v8bf*)&As[mrow * 40 + koff];
    af.h[1] = *(const v8bf*)&As[mrow * 40 + 16 + koff];
    int nrow = wn * 32 + (lane & 15);
    int kb = (lane >> 4) * 8;
    bf0.h[0] = *(const v8bf*)&Bs[nrow * 40 + kb];
    bf0.h[1] = *(const v8bf*)&Bs[nrow * 40 + kb + 16];
    bf1.h[0] = *(const v8bf*)&Bs[(nrow + 16) * 40 + kb];
    bf1.h[1] = *(const v8bf*)&Bs[(nrow + 16) * 40 + kb + 16];

    acc0 = __builtin_amdgcn_wmma_f32_16x16x32_bf16(false, af.v, false, bf0.v, (short)0, acc0, false, false);
    acc1 = __builtin_amdgcn_wmma_f32_16x16x32_bf16(false, af.v, false, bf1.v, (short)0, acc1, false, false);
    __syncthreads();
  }

  int crow = bm + wm * 16 + (lane >> 4) * 8;
  int col0 = bn + wn * 32 + (lane & 15);
  int col1 = col0 + 16;
  float bia0 = (bias && col0 < N) ? bias[col0] : 0.f;
  float bia1 = (bias && col1 < N) ? bias[col1] : 0.f;
  #pragma unroll
  for (int r = 0; r < 8; r++) {
    if (col0 < N) C[(size_t)(crow + r) * ldc + col0] = acc0[r] + bia0;
    if (col1 < N) C[(size_t)(crow + r) * ldc + col1] = acc1[r] + bia1;
  }
}

// ---------------------------------------------------------------------------
// Stage 0: L = logmap0(proj(expmap0(proj_tan0([0|embed])))) (6144 x 544 f32,
// K-padded with zeros), plus f1 = [0|embed] into d_out cols [0,513).
// ---------------------------------------------------------------------------
__global__ __launch_bounds__(256)
void stage0_kernel(const float* __restrict__ embed, float* __restrict__ L,
                   float* __restrict__ out)
{
  __shared__ float red[256];
  int j = blockIdx.x, d = threadIdx.x;
  float v1 = embed[(size_t)j * IN_DIM + d];
  float v2 = embed[(size_t)j * IN_DIM + d + 256];
  float xn = fmaxf(sqrtf(blockSum(v1 * v1 + v2 * v2, red)), MINN);
  float sh = sinhf(xn);
  float y1 = sh * v1 / xn, y2 = sh * v2 / xn;
  float sy = blockSum(y1 * y1 + y2 * y2, red);
  float x0 = sqrtf(fmaxf(1.f + sy, EPSC));
  float yn = fmaxf(sqrtf(sy), MINN);
  float fac = acoshf(fmaxf(x0, 1.f + EPSC)) / yn;
  float* Lr = L + (size_t)j * KPAD_L;
  Lr[1 + d] = fac * y1;
  Lr[1 + d + 256] = fac * y2;
  if (d == 0) Lr[0] = 0.f;
  if (d < KPAD_L - D0) Lr[D0 + d] = 0.f;
  float* orow = out + (size_t)j * OUT_COLS;
  orow[1 + d] = v1;
  orow[1 + d + 256] = v2;
  if (d == 0) orow[0] = 0.f;
}

// ub = logmap0(proj(expmap0(proj_tan0(hyp_b[c])))), class constant (256 f32).
__global__ __launch_bounds__(256)
void bias_prep_kernel(const float* __restrict__ hb, float* __restrict__ ub)
{
  __shared__ float red[256];
  int d = threadIdx.x;
  float b = hb[d];
  float xn = fmaxf(sqrtf(blockSum((d >= 1) ? b * b : 0.f, red)), MINN);
  float sh = sinhf(xn);
  float y = (d >= 1) ? sh * b / xn : 0.f;
  float sy = blockSum(y * y, red);
  float x0 = sqrtf(fmaxf(1.f + sy, EPSC));
  float yn = fmaxf(sqrtf(sy), MINN);
  float fac = acoshf(fmaxf(x0, 1.f + EPSC)) / yn;
  ub[d] = (d == 0) ? 0.f : fac * y;
}

// ---------------------------------------------------------------------------
// rowA: mv_pre -> expmap0/proj -> mobius_add(bias) -> logmap0 -> score.
// Writes Gt (bf16, TRANSPOSED [n][j], 320 x 6144): row d = score*xt_d,
// row 256 = score, rows 257..319 = 0 (pad for unguarded async staging).
// ---------------------------------------------------------------------------
__global__ __launch_bounds__(256)
void rowA_kernel(const float* __restrict__ mvpre, const float* __restrict__ ub,
                 const float* __restrict__ attw, const float* __restrict__ attb_p,
                 __bf16* __restrict__ Gt)
{
  __shared__ float red[256];
  int j = blockIdx.x, d = threadIdx.x;
  float p = mvpre[(size_t)j * HYPER + d];
  float xn = fmaxf(sqrtf(blockSum((d >= 1) ? p * p : 0.f, red)), MINN);
  float sh = sinhf(xn);
  float y = (d >= 1) ? sh * p / xn : 0.f;
  float sy = blockSum(y * y, red);
  float mv0 = sqrtf(fmaxf(1.f + sy, EPSC));
  float mv = (d == 0) ? mv0 : y;
  float yn = fmaxf(sqrtf(sy), MINN);
  float yhat = (d >= 1) ? y / yn : 0.f;
  float u = ub[d];
  float alpha = blockSum((d >= 1) ? yhat * u : 0.f, red);
  float w = (d == 0) ? alpha * yn : (u - alpha * (1.f - mv0) * yhat);
  float ux = blockSum((d >= 1) ? mv * w : 0.f, red);
  float u0 = ux / fmaxf(mv0, EPSC);
  float v2 = (d == 0) ? u0 : w;
  float dall = blockSum(v2 * v2, red);
  float md = dall - 2.f * u0 * u0;
  float normu = fminf(sqrtf(fmaxf(md, EPSC)), 1000000.f);
  float theta = fmaxf(normu, MINN);
  float ch = coshf(theta), shh = sinhf(theta);
  float h = ch * mv + shh * v2 / theta;
  float hs = blockSum((d >= 1) ? h * h : 0.f, red);
  float h0 = sqrtf(fmaxf(1.f + hs, EPSC));
  float yn2 = fmaxf(sqrtf(hs), MINN);
  float fac = acoshf(fmaxf(h0, 1.f + EPSC)) / yn2;
  float xt = (d == 0) ? 0.f : fac * h;
  float sdot = blockSum(xt * attw[d], red);
  float sc = 1.f / (1.f + expf(-(sdot + attb_p[0])));
  Gt[(size_t)d * N_NODES + j] = f2bf(sc * xt);
  if (d < GROWS_PAD - HYPER)
    Gt[(size_t)(HYPER + d) * N_NODES + j] = f2bf((d == 0) ? sc : 0.f);
}

// ---------------------------------------------------------------------------
// rowB: h_pre = Hraw[:,0:256]/(Hraw[:,256]+1e-12) -> expmap0/proj ->
// relu(logmap0) -> expmap0/proj -> proj_tan0(logmap0) = U
// ---------------------------------------------------------------------------
__global__ __launch_bounds__(256)
void rowB_kernel(const float* __restrict__ Hraw, float* __restrict__ U)
{
  __shared__ float red[256];
  int j = blockIdx.x, d = threadIdx.x;
  const float* Hr = Hraw + (size_t)j * GCOLS;
  float den = Hr[HYPER] + 1e-12f;
  float hp = Hr[d] / den;
  float xn = fmaxf(sqrtf(blockSum((d >= 1) ? hp * hp : 0.f, red)), MINN);
  float sh = sinhf(xn);
  float y = (d >= 1) ? sh * hp / xn : 0.f;
  float sy = blockSum(y * y, red);
  float h0 = sqrtf(fmaxf(1.f + sy, EPSC));
  float yn = fmaxf(sqrtf(sy), MINN);
  float fac = acoshf(fmaxf(h0, 1.f + EPSC)) / yn;
  float r = fmaxf((d == 0) ? 0.f : fac * y, 0.f);
  float xn2 = fmaxf(sqrtf(blockSum((d >= 1) ? r * r : 0.f, red)), MINN);
  float sh2 = sinhf(xn2);
  float y2 = (d >= 1) ? sh2 * r / xn2 : 0.f;
  float sy2 = blockSum(y2 * y2, red);
  float h20 = sqrtf(fmaxf(1.f + sy2, EPSC));
  float yn3 = fmaxf(sqrtf(sy2), MINN);
  float fac3 = acoshf(fmaxf(h20, 1.f + EPSC)) / yn3;
  U[(size_t)j * HYPER + d] = (d == 0) ? 0.f : fac3 * y2;
}

__global__ __launch_bounds__(256)
void copy_kernel(const float* __restrict__ src, float* __restrict__ dst, int n)
{
  int i = blockIdx.x * blockDim.x + threadIdx.x;
  if (i < n) dst[i] = src[i];
}

// z[dst[e]] += T[src[e]] : hardware f32 atomics (global_atomic_add_f32)
__global__ __launch_bounds__(256)
void scatter_kernel(const float* __restrict__ T, const int* __restrict__ srcI,
                    const int* __restrict__ dstI, float* __restrict__ Z)
{
  int e = blockIdx.x;
  int s = srcI[e], dn = dstI[e];
  int d = threadIdx.x;
  unsafeAtomicAdd(&Z[(size_t)dn * HYPER + d], T[(size_t)s * HYPER + d]);
}

__global__ __launch_bounds__(256)
void bn_stats_kernel(const float* __restrict__ Z, float* __restrict__ mu,
                     float* __restrict__ var)
{
  __shared__ float red[256];
  int d = blockIdx.x, t = threadIdx.x;
  float s = 0.f, sq = 0.f;
  for (int r = t; r < N_NODES; r += 256) {
    float v = Z[(size_t)r * HYPER + d];
    s += v; sq += v * v;
  }
  float ts = blockSum(s, red);
  float tq = blockSum(sq, red);
  if (t == 0) {
    float m = ts / (float)N_NODES;
    mu[d] = m;
    var[d] = tq / (float)N_NODES - m * m;
  }
}

__global__ __launch_bounds__(256)
void bn_apply_kernel(float* __restrict__ Z, const float* __restrict__ mu,
                     const float* __restrict__ var, const float* __restrict__ g,
                     const float* __restrict__ b)
{
  int j = blockIdx.x, d = threadIdx.x;
  size_t idx = (size_t)j * HYPER + d;
  float v = Z[idx];
  Z[idx] = tanhf(g[d] * (v - mu[d]) * rsqrtf(var[d] + 1e-5f) + b[d]);
}

// ---------------------------------------------------------------------------
extern "C" void kernel_launch(void* const* d_in, const int* in_sizes, int n_in,
                              void* d_out, int out_size, void* d_ws, size_t ws_size,
                              hipStream_t stream)
{
  const float* embed  = (const float*)d_in[0];
  const float* adj    = (const float*)d_in[1];
  const float* hyp_W  = (const float*)d_in[2];
  const float* hyp_b  = (const float*)d_in[3];
  const float* att_w  = (const float*)d_in[4];
  const float* att_b  = (const float*)d_in[5];
  const float* dec_W  = (const float*)d_in[6];
  const float* dec_b  = (const float*)d_in[7];
  const float* gin_W1 = (const float*)d_in[8];
  const float* gin_b1 = (const float*)d_in[9];
  const float* bn_g   = (const float*)d_in[10];
  const float* bn_b   = (const float*)d_in[11];
  const float* gin_W2 = (const float*)d_in[12];
  const float* gin_b2 = (const float*)d_in[13];
  const int*   eidx   = (const int*)d_in[14];
  const int* esrc = eidx;
  const int* edst = eidx + N_EDGES;
  float* out = (float*)d_out;

  // workspace layout (~118 MB, buffers aliased across pipeline stages)
  char* base = (char*)d_ws;
  __bf16* adjb = (__bf16*)base; base += (size_t)N_NODES * N_NODES * 2;
  __bf16* Gt   = (__bf16*)base; base += (size_t)GROWS_PAD * N_NODES * 2;
  float* L     = (float*)base;  base += (size_t)N_NODES * KPAD_L * 4;
  float* Hraw  = (float*)base;  base += (size_t)N_NODES * GCOLS * 4;
  float* bufA  = (float*)base;  base += (size_t)N_NODES * HYPER * 4; // mvpre/U/Z1
  float* bufB  = (float*)base;  base += (size_t)N_NODES * HYPER * 4; // T
  float* bufC  = (float*)base;  base += (size_t)N_NODES * HYPER * 4; // Z
  float* ub    = (float*)base;  base += 256 * 4;
  float* mu    = (float*)base;  base += 256 * 4;
  float* var   = (float*)base;  base += 256 * 4;

  cvt_adj_kernel<<<(size_t)N_NODES * N_NODES / (4 * 256), 256, 0, stream>>>(adj, adjb);
  stage0_kernel<<<N_NODES, 256, 0, stream>>>(embed, L, out);

  for (int c = 0; c < C_CLASS; c++) {
    bias_prep_kernel<<<1, 256, 0, stream>>>(hyp_b + c * HYPER, ub);

    // mv_pre = L @ hyp_W[c].T   (M=6144, N=256, K=513 padded to 544)
    gemm_wmma_f32in<<<dim3(HYPER / 64, N_NODES / 64), 256, 0, stream>>>(
        L, KPAD_L, hyp_W + (size_t)c * HYPER * D0, D0, nullptr,
        bufA, HYPER, HYPER, KPAD_L, D0);

    rowA_kernel<<<N_NODES, 256, 0, stream>>>(bufA, ub, att_w + c * HYPER,
                                             att_b + c, Gt);

    // Hraw = adjb @ Gt^T  (M=6144, N=272, K=6144; pure-bf16 async-LDS GEMM)
    gemm_wmma_bf16bf16<<<dim3((GCOLS + 63) / 64, N_NODES / 128), 256, 0, stream>>>(
        adjb, N_NODES, Gt, N_NODES, Hraw, GCOLS, GCOLS, N_NODES);

    rowB_kernel<<<N_NODES, 256, 0, stream>>>(Hraw, bufA);   // -> U

    // T = U @ dec_W[c].T + dec_b[c]
    gemm_wmma_f32in<<<dim3(HYPER / 64, N_NODES / 64), 256, 0, stream>>>(
        bufA, HYPER, dec_W + (size_t)c * HYPER * HYPER, HYPER, dec_b + c * HYPER,
        bufB, HYPER, HYPER, HYPER, HYPER);

    // Z = T + segment_sum(T[src], dst)
    copy_kernel<<<(N_NODES * HYPER) / 256, 256, 0, stream>>>(bufB, bufC, N_NODES * HYPER);
    scatter_kernel<<<N_EDGES, 256, 0, stream>>>(bufB, esrc, edst, bufC);

    // Z1 = Z @ gin_W1[c].T + gin_b1[c]
    gemm_wmma_f32in<<<dim3(HYPER / 64, N_NODES / 64), 256, 0, stream>>>(
        bufC, HYPER, gin_W1 + (size_t)c * HYPER * HYPER, HYPER, gin_b1 + c * HYPER,
        bufA, HYPER, HYPER, HYPER, HYPER);

    // batchnorm + tanh (in place)
    bn_stats_kernel<<<HYPER, 256, 0, stream>>>(bufA, mu, var);
    bn_apply_kernel<<<N_NODES, 256, 0, stream>>>(bufA, mu, var, bn_g + c * HYPER,
                                                 bn_b + c * HYPER);

    // out block = Z1 @ gin_W2[c].T + gin_b2[c], written directly into d_out
    gemm_wmma_f32in<<<dim3(HYPER / 64, N_NODES / 64), 256, 0, stream>>>(
        bufA, HYPER, gin_W2 + (size_t)c * HYPER * HYPER, HYPER, gin_b2 + c * HYPER,
        out + D0 + (size_t)c * HYPER, OUT_COLS, HYPER, HYPER, HYPER);
  }
}
